// TransformerBlock_45251775430985
// MI455X (gfx1250) — compile-verified, matrix operand order checked by reading, not run on hardware
//
#include <hip/hip_runtime.h>
#include <hip/hip_bf16.h>

#define T_  4096
#define D_  768
#define H_  12
#define HD_ 64
#define FF_ 3072
#define EPS_ 1e-5f

#define BM 64
#define BN 128
#define BK 32
#define SAS 48   // padded LDS row stride (elements) for A tile
#define SBS 48   // padded LDS row stride (elements) for B tile

#define USE_ASYNC_COPY 1   // gfx1250 GLOBAL_LOAD_ASYNC_TO_LDS path (ASYNCcnt)

typedef __attribute__((ext_vector_type(16))) __bf16          v16bf;
typedef __attribute__((ext_vector_type(8)))  float           v8f;
typedef __attribute__((ext_vector_type(8)))  unsigned short  us8;

__device__ __forceinline__ unsigned short f2bf(float f) {
    unsigned int u = __builtin_bit_cast(unsigned int, f);
    u += 0x7FFFu + ((u >> 16) & 1u);            // round-to-nearest-even
    return (unsigned short)(u >> 16);
}

// Load one 16-element bf16 fragment as two contiguous 16-byte vectors.
__device__ __forceinline__ v16bf ldfrag(const unsigned short* p, int base0, int base1) {
    union { us8 h[2]; v16bf v; } fc;
    fc.h[0] = *(const us8*)(p + base0);
    fc.h[1] = *(const us8*)(p + base1);
    return fc.v;
}

#if USE_ASYNC_COPY
// GVS-mode async copy: LDS[ldsAddr] = MEM[saddr + byteOff], 16 bytes per lane.
// Tracked by ASYNCcnt (not LOADcnt/DScnt).
__device__ __forceinline__ void async_copy_b128(const void* gbase, unsigned byteOff,
                                                unsigned ldsAddr) {
    asm volatile("global_load_async_to_lds_b128 %0, %1, %2 offset:0"
                 :: "v"(ldsAddr), "v"(byteOff), "s"(gbase)
                 : "memory");
}
__device__ __forceinline__ void wait_async0() {
    asm volatile("s_wait_asynccnt 0x0" ::: "memory");
}
// low 32 bits of a flat pointer into LDS are the LDS byte address
__device__ __forceinline__ unsigned lds_addr(const void* p) {
    return (unsigned)(size_t)p;
}
#endif

// ---------------- fp32 -> bf16 convert + transpose: in[K][N] -> out[N][K] ----------------
__global__ __launch_bounds__(256)
void convert_transpose_bf16(const float* __restrict__ in, unsigned short* __restrict__ out,
                            int K, int N) {
    int i = blockIdx.x * 256 + threadIdx.x;
    if (i < K * N) {
        int k = i / N, n = i % N;
        out[(size_t)n * K + k] = f2bf(in[i]);
    }
}

// ---------------- layernorm (+ cast to bf16) ----------------
__global__ __launch_bounds__(256)
void layernorm_cast(const float* __restrict__ x, const float* __restrict__ g,
                    const float* __restrict__ b, unsigned short* __restrict__ out) {
    int row = blockIdx.x, tid = threadIdx.x;
    int lane = tid & 31, wave = tid >> 5;
    const float* xr = x + (size_t)row * D_;
    float sum = 0.f, sq = 0.f;
    for (int i = tid; i < D_; i += 256) { float v = xr[i]; sum += v; sq += v * v; }
    for (int off = 16; off > 0; off >>= 1) {
        sum += __shfl_xor(sum, off, 32);
        sq  += __shfl_xor(sq,  off, 32);
    }
    __shared__ float ssum[8], ssq[8];
    if (lane == 0) { ssum[wave] = sum; ssq[wave] = sq; }
    __syncthreads();
    float ts = 0.f, tq = 0.f;
    for (int w = 0; w < 8; w++) { ts += ssum[w]; tq += ssq[w]; }
    float mean = ts / (float)D_;
    float var  = tq / (float)D_ - mean * mean;
    float rstd = rsqrtf(var + EPS_);
    for (int i = tid; i < D_; i += 256) {
        float v = (xr[i] - mean) * rstd * g[i] + b[i];
        out[(size_t)row * D_ + i] = f2bf(v);
    }
}

// ---------------- bf16 GEMM (WMMA): C = A[M,K] @ Bt[N,K]^T ----------------
// EPI 0: Cout = bf16(acc)
// EPI 1: Cout = f32(acc + bias[col] + res[row,col])
template <int EPI>
__global__ __launch_bounds__(256)
void gemm_bf16_wmma(const unsigned short* __restrict__ A,    // [M,K] bf16 row-major
                    const unsigned short* __restrict__ Bt,   // [N,K] bf16 (pre-transposed)
                    const float* __restrict__ bias,
                    const float* __restrict__ res,
                    void* __restrict__ Cout,
                    int M, int N, int K) {
    __shared__ unsigned short sA[BM * SAS];
    __shared__ unsigned short sB[BN * SBS];
    int tid = threadIdx.x;
    int wave = tid >> 5, lane = tid & 31, lh = lane >> 4, l16 = lane & 15;
    int wm = wave & 1, wn = wave >> 1;           // 2 x 4 wave grid
    int m0 = blockIdx.x * BM, n0 = blockIdx.y * BN;

    v8f acc[2][2];
#pragma unroll
    for (int i = 0; i < 2; i++)
#pragma unroll
        for (int j = 0; j < 2; j++)
#pragma unroll
            for (int e = 0; e < 8; e++) acc[i][j][e] = 0.0f;

    const int ar = tid >> 2, ac = (tid & 3) * 8;   // A: 64 rows x 4 threads x 8 elems
    const int rb = tid >> 1, cb = (tid & 1) * 16;  // Bt: 128 rows x 2 threads x 16 elems
    const int base0 = 8 * lh, base1 = 16 + 8 * lh;
#if USE_ASYNC_COPY
    const unsigned ldsA  = lds_addr(&sA[ar * SAS + ac]);
    const unsigned ldsB0 = lds_addr(&sB[rb * SBS + cb]);
    const size_t gA  = ((size_t)(m0 + ar) * K + ac) * 2;
    const size_t gB  = ((size_t)(n0 + rb) * K + cb) * 2;
#endif

    for (int k0 = 0; k0 < K; k0 += BK) {
#if USE_ASYNC_COPY
        async_copy_b128(A,  (unsigned)(gA + (size_t)k0 * 2),      ldsA);
        async_copy_b128(Bt, (unsigned)(gB + (size_t)k0 * 2),      ldsB0);
        async_copy_b128(Bt, (unsigned)(gB + (size_t)k0 * 2 + 16), ldsB0 + 16);
        wait_async0();
#else
        *(us8*)&sA[ar * SAS + ac]     = *(const us8*)&A[(size_t)(m0 + ar) * K + k0 + ac];
        *(us8*)&sB[rb * SBS + cb]     = *(const us8*)&Bt[(size_t)(n0 + rb) * K + k0 + cb];
        *(us8*)&sB[rb * SBS + cb + 8] = *(const us8*)&Bt[(size_t)(n0 + rb) * K + k0 + cb + 8];
#endif
        if (k0 + BK < K) {
            __builtin_prefetch(&A[(size_t)(m0 + ar) * K + k0 + BK + ac], 0, 0);
            __builtin_prefetch(&Bt[(size_t)(n0 + rb) * K + k0 + BK + cb], 0, 0);
        }
        __syncthreads();

        v16bf af[2], bfrag[2];
#pragma unroll
        for (int mi = 0; mi < 2; mi++)
            af[mi] = ldfrag(&sA[(wm * 32 + mi * 16 + l16) * SAS], base0, base1);
#pragma unroll
        for (int ni = 0; ni < 2; ni++)
            bfrag[ni] = ldfrag(&sB[(wn * 32 + ni * 16 + l16) * SBS], base0, base1);
#pragma unroll
        for (int mi = 0; mi < 2; mi++)
#pragma unroll
            for (int ni = 0; ni < 2; ni++)
                acc[mi][ni] = __builtin_amdgcn_wmma_f32_16x16x32_bf16(
                    false, af[mi], false, bfrag[ni], (short)0, acc[mi][ni], false, false);
        __syncthreads();
    }

#pragma unroll
    for (int mi = 0; mi < 2; mi++) {
        int rbase = m0 + wm * 32 + mi * 16 + 8 * lh;
#pragma unroll
        for (int ni = 0; ni < 2; ni++) {
            int col = n0 + wn * 32 + ni * 16 + l16;
#pragma unroll
            for (int r = 0; r < 8; r++) {
                int row = rbase + r;
                float v = acc[mi][ni][r];
                if (EPI == 0) {
                    ((unsigned short*)Cout)[(size_t)row * N + col] = f2bf(v);
                } else {
                    ((float*)Cout)[(size_t)row * N + col] =
                        v + bias[col] + res[(size_t)row * N + col];
                }
            }
        }
    }
}

// ---------------- dual-weight gated GEMM: bf16( silu(A@B1t^T+b1) * (A@B2t^T+b2) ) ----------------
__global__ __launch_bounds__(256)
void gemm_gated_wmma(const unsigned short* __restrict__ A,     // [M,K]
                     const unsigned short* __restrict__ B1t,   // [N,K]
                     const unsigned short* __restrict__ B2t,   // [N,K]
                     const float* __restrict__ bias1,
                     const float* __restrict__ bias2,
                     unsigned short* __restrict__ Cout,
                     int M, int N, int K) {
    __shared__ unsigned short sA[BM * SAS];
    __shared__ unsigned short sB1[BN * SBS];
    __shared__ unsigned short sB2[BN * SBS];
    int tid = threadIdx.x;
    int wave = tid >> 5, lane = tid & 31, lh = lane >> 4, l16 = lane & 15;
    int wm = wave & 1, wn = wave >> 1;
    int m0 = blockIdx.x * BM, n0 = blockIdx.y * BN;

    v8f acc1[2][2], acc2[2][2];
#pragma unroll
    for (int i = 0; i < 2; i++)
#pragma unroll
        for (int j = 0; j < 2; j++)
#pragma unroll
            for (int e = 0; e < 8; e++) { acc1[i][j][e] = 0.0f; acc2[i][j][e] = 0.0f; }

    const int ar = tid >> 2, ac = (tid & 3) * 8;
    const int rb = tid >> 1, cb = (tid & 1) * 16;
    const int base0 = 8 * lh, base1 = 16 + 8 * lh;
#if USE_ASYNC_COPY
    const unsigned ldsA  = lds_addr(&sA[ar * SAS + ac]);
    const unsigned ldsB1 = lds_addr(&sB1[rb * SBS + cb]);
    const unsigned ldsB2 = lds_addr(&sB2[rb * SBS + cb]);
    const size_t gA = ((size_t)(m0 + ar) * K + ac) * 2;
    const size_t gB = ((size_t)(n0 + rb) * K + cb) * 2;
#endif

    for (int k0 = 0; k0 < K; k0 += BK) {
#if USE_ASYNC_COPY
        async_copy_b128(A,   (unsigned)(gA + (size_t)k0 * 2),      ldsA);
        async_copy_b128(B1t, (unsigned)(gB + (size_t)k0 * 2),      ldsB1);
        async_copy_b128(B1t, (unsigned)(gB + (size_t)k0 * 2 + 16), ldsB1 + 16);
        async_copy_b128(B2t, (unsigned)(gB + (size_t)k0 * 2),      ldsB2);
        async_copy_b128(B2t, (unsigned)(gB + (size_t)k0 * 2 + 16), ldsB2 + 16);
        wait_async0();
#else
        *(us8*)&sA[ar * SAS + ac]      = *(const us8*)&A[(size_t)(m0 + ar) * K + k0 + ac];
        *(us8*)&sB1[rb * SBS + cb]     = *(const us8*)&B1t[(size_t)(n0 + rb) * K + k0 + cb];
        *(us8*)&sB1[rb * SBS + cb + 8] = *(const us8*)&B1t[(size_t)(n0 + rb) * K + k0 + cb + 8];
        *(us8*)&sB2[rb * SBS + cb]     = *(const us8*)&B2t[(size_t)(n0 + rb) * K + k0 + cb];
        *(us8*)&sB2[rb * SBS + cb + 8] = *(const us8*)&B2t[(size_t)(n0 + rb) * K + k0 + cb + 8];
#endif
        __syncthreads();

        v16bf af[2];
#pragma unroll
        for (int mi = 0; mi < 2; mi++)
            af[mi] = ldfrag(&sA[(wm * 32 + mi * 16 + l16) * SAS], base0, base1);
#pragma unroll
        for (int ni = 0; ni < 2; ni++) {
            int c = wn * 32 + ni * 16 + l16;
            v16bf b1f = ldfrag(&sB1[c * SBS], base0, base1);
            v16bf b2f = ldfrag(&sB2[c * SBS], base0, base1);
#pragma unroll
            for (int mi = 0; mi < 2; mi++) {
                acc1[mi][ni] = __builtin_amdgcn_wmma_f32_16x16x32_bf16(
                    false, af[mi], false, b1f, (short)0, acc1[mi][ni], false, false);
                acc2[mi][ni] = __builtin_amdgcn_wmma_f32_16x16x32_bf16(
                    false, af[mi], false, b2f, (short)0, acc2[mi][ni], false, false);
            }
        }
        __syncthreads();
    }

#pragma unroll
    for (int mi = 0; mi < 2; mi++) {
        int rbase = m0 + wm * 32 + mi * 16 + 8 * lh;
#pragma unroll
        for (int ni = 0; ni < 2; ni++) {
            int col = n0 + wn * 32 + ni * 16 + l16;
#pragma unroll
            for (int r = 0; r < 8; r++) {
                int row = rbase + r;
                float gv = acc1[mi][ni][r] + bias1[col];
                float uv = acc2[mi][ni][r] + bias2[col];
                float y  = (gv / (1.0f + __expf(-gv))) * uv;   // silu(gv)*uv
                Cout[(size_t)row * N + col] = f2bf(y);
            }
        }
    }
}

// ---------------- V transpose: [T, H*HD] -> [H*HD, T] ----------------
__global__ __launch_bounds__(256)
void transpose_v(const unsigned short* __restrict__ v, unsigned short* __restrict__ vt) {
    int i = blockIdx.x * 256 + threadIdx.x;
    if (i < T_ * D_) {
        int t = i / D_, c = i % D_;
        vt[(size_t)c * T_ + t] = v[i];
    }
}

// ---------------- causal flash attention (WMMA) ----------------
// Q,K: [T, H*HD] bf16 ; Vt: [H*HD, T] bf16 ; O: [T, H*HD] bf16
__global__ __launch_bounds__(256)
void flash_attn(const unsigned short* __restrict__ Q,
                const unsigned short* __restrict__ Kc,
                const unsigned short* __restrict__ Vt,
                unsigned short* __restrict__ O) {
    __shared__ unsigned short sP[8][16 * 32];   // per-wave P transpose staging
    int h = blockIdx.x;
    int wave = threadIdx.x >> 5, lane = threadIdx.x & 31;
    int lh = lane >> 4, l16 = lane & 15;
    int q0 = blockIdx.y * 128 + wave * 16;
    unsigned short* sPw = sP[wave];
    const int base0 = 8 * lh, base1 = 16 + 8 * lh;

    const unsigned short* qp = Q + (size_t)(q0 + l16) * D_ + h * HD_;
    v16bf qf[2];
#pragma unroll
    for (int f = 0; f < 2; f++) qf[f] = ldfrag(qp + f * 32, base0, base1);

    v8f ctx[4];
#pragma unroll
    for (int t = 0; t < 4; t++)
#pragma unroll
        for (int e = 0; e < 8; e++) ctx[t][e] = 0.0f;
    float mrow[8], lrow[8];
#pragma unroll
    for (int r = 0; r < 8; r++) { mrow[r] = -1e30f; lrow[r] = 0.0f; }

    const float scale = 0.125f;                 // 1/sqrt(HD=64)
    const int kend = q0 + 16;                   // causal bound for this wave
    for (int kb = 0; kb < kend; kb += 32) {
        v8f S[2];
#pragma unroll
        for (int nt = 0; nt < 2; nt++)
#pragma unroll
            for (int e = 0; e < 8; e++) S[nt][e] = 0.0f;
#pragma unroll
        for (int nt = 0; nt < 2; nt++) {
            const unsigned short* kp = Kc + (size_t)(kb + nt * 16 + l16) * D_ + h * HD_;
#pragma unroll
            for (int f = 0; f < 2; f++) {
                v16bf kf = ldfrag(kp + f * 32, base0, base1);
                S[nt] = __builtin_amdgcn_wmma_f32_16x16x32_bf16(
                    false, qf[f], false, kf, (short)0, S[nt], false, false);
            }
        }

        const int c0 = kb + l16, c1 = kb + 16 + l16;
#pragma unroll
        for (int r = 0; r < 8; r++) {
            int row = q0 + r + 8 * lh;
            float s0 = (c0 <= row) ? S[0][r] * scale : -1e30f;
            float s1 = (c1 <= row) ? S[1][r] * scale : -1e30f;
            float mx = fmaxf(s0, s1);
#pragma unroll
            for (int off = 1; off < 16; off <<= 1) mx = fmaxf(mx, __shfl_xor(mx, off, 32));
            float mn = fmaxf(mrow[r], mx);
            float p0 = __expf(s0 - mn), p1 = __expf(s1 - mn);
            float rs = p0 + p1;
#pragma unroll
            for (int off = 1; off < 16; off <<= 1) rs += __shfl_xor(rs, off, 32);
            float corr = __expf(mrow[r] - mn);
            lrow[r] = lrow[r] * corr + rs;
            mrow[r] = mn;
#pragma unroll
            for (int t = 0; t < 4; t++) ctx[t][r] *= corr;
            sPw[(r + 8 * lh) * 32 + l16]      = f2bf(p0);
            sPw[(r + 8 * lh) * 32 + 16 + l16] = f2bf(p1);
        }

        v16bf pf = ldfrag(&sPw[l16 * 32], base0, base1);

#pragma unroll
        for (int t = 0; t < 4; t++) {
            const unsigned short* vp = Vt + (size_t)(h * HD_ + t * 16 + l16) * T_ + kb;
            v16bf vf = ldfrag(vp, base0, base1);
            ctx[t] = __builtin_amdgcn_wmma_f32_16x16x32_bf16(
                false, pf, false, vf, (short)0, ctx[t], false, false);
        }
    }

#pragma unroll
    for (int t = 0; t < 4; t++) {
        int hd = t * 16 + l16;
#pragma unroll
        for (int r = 0; r < 8; r++) {
            int row = q0 + r + 8 * lh;
            O[(size_t)row * D_ + h * HD_ + hd] = f2bf(ctx[t][r] / lrow[r]);
        }
    }
}

// ---------------- launcher ----------------
extern "C" void kernel_launch(void* const* d_in, const int* in_sizes, int n_in,
                              void* d_out, int out_size, void* d_ws, size_t ws_size,
                              hipStream_t stream) {
    (void)in_sizes; (void)n_in; (void)out_size; (void)ws_size;
    const float* x  = (const float*)d_in[0];
    const float* Wq = (const float*)d_in[1];
    const float* Wk = (const float*)d_in[2];
    const float* Wv = (const float*)d_in[3];
    const float* Wo = (const float*)d_in[4];
    const float* bo = (const float*)d_in[5];
    const float* w1 = (const float*)d_in[6];
    const float* b1 = (const float*)d_in[7];
    const float* w2 = (const float*)d_in[8];
    const float* b2 = (const float*)d_in[9];
    const float* w3 = (const float*)d_in[10];
    const float* b3 = (const float*)d_in[11];
    const float* g1 = (const float*)d_in[12];
    const float* s1 = (const float*)d_in[13];
    const float* g2 = (const float*)d_in[14];
    const float* s2 = (const float*)d_in[15];

    char* wsb = (char*)d_ws;
    size_t off = 0;
    auto take = [&](size_t bytes) -> void* {
        void* p = wsb + off;
        off += (bytes + 255) & ~(size_t)255;
        return p;
    };

    const size_t DD = (size_t)D_ * D_;
    const size_t DF = (size_t)D_ * FF_;
    const size_t TD = (size_t)T_ * D_;
    const size_t TF = (size_t)T_ * FF_;

    unsigned short* wq_t  = (unsigned short*)take(DD * 2);   // [N=768][K=768]
    unsigned short* wk_t  = (unsigned short*)take(DD * 2);
    unsigned short* wv_t  = (unsigned short*)take(DD * 2);
    unsigned short* wo_t  = (unsigned short*)take(DD * 2);
    unsigned short* w1_t  = (unsigned short*)take(DF * 2);   // [N=3072][K=768]
    unsigned short* w2_t  = (unsigned short*)take(DF * 2);
    unsigned short* w3_t  = (unsigned short*)take(DF * 2);   // [N=768][K=3072]
    unsigned short* h1_bf = (unsigned short*)take(TD * 2);
    unsigned short* q_bf  = (unsigned short*)take(TD * 2);
    unsigned short* k_bf  = (unsigned short*)take(TD * 2);
    unsigned short* v_bf  = (unsigned short*)take(TD * 2);
    unsigned short* vt_bf = (unsigned short*)take(TD * 2);
    unsigned short* ctx_bf= (unsigned short*)take(TD * 2);
    float*          x1    = (float*)take(TD * 4);
    unsigned short* h2_bf = (unsigned short*)take(TD * 2);
    unsigned short* gate_bf=(unsigned short*)take(TF * 2);

    convert_transpose_bf16<<<(int)((DD + 255) / 256), 256, 0, stream>>>(Wq, wq_t, D_, D_);
    convert_transpose_bf16<<<(int)((DD + 255) / 256), 256, 0, stream>>>(Wk, wk_t, D_, D_);
    convert_transpose_bf16<<<(int)((DD + 255) / 256), 256, 0, stream>>>(Wv, wv_t, D_, D_);
    convert_transpose_bf16<<<(int)((DD + 255) / 256), 256, 0, stream>>>(Wo, wo_t, D_, D_);
    convert_transpose_bf16<<<(int)((DF + 255) / 256), 256, 0, stream>>>(w1, w1_t, D_, FF_);
    convert_transpose_bf16<<<(int)((DF + 255) / 256), 256, 0, stream>>>(w2, w2_t, D_, FF_);
    convert_transpose_bf16<<<(int)((DF + 255) / 256), 256, 0, stream>>>(w3, w3_t, FF_, D_);

    layernorm_cast<<<T_, 256, 0, stream>>>(x, g1, s1, h1_bf);

    dim3 gDD(T_ / BM, D_ / BN);
    gemm_bf16_wmma<0><<<gDD, 256, 0, stream>>>(h1_bf, wq_t, nullptr, nullptr, q_bf, T_, D_, D_);
    gemm_bf16_wmma<0><<<gDD, 256, 0, stream>>>(h1_bf, wk_t, nullptr, nullptr, k_bf, T_, D_, D_);
    gemm_bf16_wmma<0><<<gDD, 256, 0, stream>>>(h1_bf, wv_t, nullptr, nullptr, v_bf, T_, D_, D_);

    transpose_v<<<(int)((TD + 255) / 256), 256, 0, stream>>>(v_bf, vt_bf);

    flash_attn<<<dim3(H_, T_ / 128), 256, 0, stream>>>(q_bf, k_bf, vt_bf, ctx_bf);

    gemm_bf16_wmma<1><<<gDD, 256, 0, stream>>>(ctx_bf, wo_t, bo, x, x1, T_, D_, D_);

    layernorm_cast<<<T_, 256, 0, stream>>>(x1, g2, s2, h2_bf);

    gemm_gated_wmma<<<dim3(T_ / BM, FF_ / BN), 256, 0, stream>>>(
        h2_bf, w1_t, w2_t, b1, b2, gate_bf, T_, FF_, D_);

    gemm_bf16_wmma<1><<<gDD, 256, 0, stream>>>(gate_bf, w3_t, b3, x1, (float*)d_out, T_, D_, FF_);
}
